// RhoVanillaVAE_35218731827918
// MI455X (gfx1250) — matrix operand order repaired and checked by reading, not compiled, not run on hardware
//
#include <hip/hip_runtime.h>
#include <math.h>

// Problem constants (match reference)
#define NB   16384   // batch rows
#define DD   4096    // DATA_DIM
#define ZD   512     // Z_DIM
#define HH   400     // H

typedef __attribute__((ext_vector_type(16))) __bf16 v16bf;
typedef __attribute__((ext_vector_type(8)))  __bf16 bf16x8;
typedef __attribute__((ext_vector_type(8)))  float  v8f;

static __device__ __forceinline__ __bf16 f2bf(float f) {
  unsigned u = __builtin_bit_cast(unsigned, f);
  u += 0x7FFFu + ((u >> 16) & 1u);                // round-to-nearest-even
  unsigned short h = (unsigned short)(u >> 16);
  return __builtin_bit_cast(__bf16, h);
}
static __device__ __forceinline__ float bf2f(__bf16 b) {
  unsigned short h = __builtin_bit_cast(unsigned short, b);
  unsigned u = ((unsigned)h) << 16;
  return __builtin_bit_cast(float, u);
}

union AFragU { v16bf f; bf16x8 h[2]; };

// ---------------------------------------------------------------------------
// elementwise f32 -> bf16 (x pre-conversion); exact sizing, 8 elems/thread
// ---------------------------------------------------------------------------
__global__ __launch_bounds__(256) void cvt_bf16_kernel(
    const float* __restrict__ src, __bf16* __restrict__ dst)
{
  size_t i = ((size_t)blockIdx.x * 256 + threadIdx.x) * 8;
  float4 a = *(const float4*)(src + i);
  float4 b = *(const float4*)(src + i + 4);
  bf16x8 o;
  o[0] = f2bf(a.x); o[1] = f2bf(a.y); o[2] = f2bf(a.z); o[3] = f2bf(a.w);
  o[4] = f2bf(b.x); o[5] = f2bf(b.y); o[6] = f2bf(b.z); o[7] = f2bf(b.w);
  *(bf16x8*)(dst + i) = o;
}

// ---------------------------------------------------------------------------
// W[R][C] f32  ->  WT[C][R] bf16   (16x16 tiles; R,C multiples of 16)
// ---------------------------------------------------------------------------
__global__ __launch_bounds__(256) void transpose_cvt_kernel(
    const float* __restrict__ W, __bf16* __restrict__ WT, int R, int C)
{
  __shared__ float t[16][17];
  const int tx = threadIdx.x & 15, ty = threadIdx.x >> 4;
  const int c0 = blockIdx.x * 16, r0 = blockIdx.y * 16;
  t[ty][tx] = W[(size_t)(r0 + ty) * C + (c0 + tx)];
  __syncthreads();
  WT[(size_t)(c0 + ty) * R + (r0 + tx)] = f2bf(t[tx][ty]);
}

// ---------------------------------------------------------------------------
// Tiled WMMA GEMM:  C = act(A[M,K](bf16) @ BT[N,K](bf16)^T + bias)
//   EPI = 0: relu -> bf16 store; 1: sigmoid -> f32 store; 2: identity -> f32
// Block tile 128x64, K-step 32, 256 threads (8 waves), wave tile 32x32.
// Ping-pong LDS double buffering, one barrier per K-step; steady-state loop
// has zero predicates (K tail handled once; B rows clamped to N-1, which is
// safe because column n of the output depends only on B row n and columns
// >= N are never stored).
// ---------------------------------------------------------------------------
template <int EPI>
__global__ __launch_bounds__(256) void gemm_bias_act(
    const __bf16* __restrict__ A, int lda,
    const __bf16* __restrict__ BT,           // [N][K], row stride = K
    const float* __restrict__ bias,
    void* __restrict__ Cp, int ldc,
    int M, int N, int K)
{
  constexpr int BM = 128, BN = 64, BK = 32;
  constexpr int LA = BK + 8;   // 40 bf16 row stride: conflict-free b128 frag loads
  constexpr int LB = BK + 8;
  __shared__ __bf16 As[2][BM * LA];   // [m][k], ping-pong
  __shared__ __bf16 Bs[2][BN * LB];   // [n][k], ping-pong

  const int tid   = threadIdx.x;
  const int lane  = tid & 31;
  const int wave  = tid >> 5;
  const int wm    = wave & 3;      // 4 wave-rows of 32
  const int wn    = wave >> 2;     // 2 wave-cols of 32
  const int lhalf = lane >> 4;     // 0/1 lane half
  const int l15   = lane & 15;
  const int blockM = blockIdx.x * BM;
  const int blockN = blockIdx.y * BN;

  // staging coordinates (hoisted; no guards needed in steady state)
  const int srow = tid >> 2;            // 0..63
  const int sc8  = (tid & 3) * 8;       // 0,8,16,24
  const size_t aRow0 = (size_t)(blockM + srow) * lda;
  const size_t aRow1 = (size_t)(blockM + srow + 64) * lda;
  int gnB = blockN + srow;
  if (gnB > N - 1) gnB = N - 1;         // clamp: cols >= N are never stored
  const size_t bRow = (size_t)gnB * K;

  bf16x8 ra0, ra1, rb;                  // staging registers (global -> LDS)
  auto gload = [&](int gk) {
    ra0 = *(const bf16x8*)(A + aRow0 + gk);
    ra1 = *(const bf16x8*)(A + aRow1 + gk);
    rb  = *(const bf16x8*)(BT + bRow + gk);
  };
  auto sstore = [&](int buf) {
    *(bf16x8*)&As[buf][srow * LA + sc8]        = ra0;
    *(bf16x8*)&As[buf][(srow + 64) * LA + sc8] = ra1;
    *(bf16x8*)&Bs[buf][srow * LB + sc8]        = rb;
  };

  v8f acc[2][2] = {};
  auto compute = [&](int buf) {
    AFragU a[2], b[2];
#pragma unroll
    for (int ti = 0; ti < 2; ++ti) {
      // A 16x32 bf16: lanes 0-15 row=l, K{0..7,16..23}; lanes 16-31 K{8..15,24..31}
      const __bf16* pa = &As[buf][(wm * 32 + ti * 16 + l15) * LA];
      a[ti].h[0] = *(const bf16x8*)(pa + lhalf * 8);
      a[ti].h[1] = *(const bf16x8*)(pa + 16 + lhalf * 8);
    }
#pragma unroll
    for (int tj = 0; tj < 2; ++tj) {
      // B 32x16 bf16: lanes 0-15 K{0..7,16..23}; lanes 16-31 K{8..15,24..31}, col = lane&15
      const __bf16* pb = &Bs[buf][(wn * 32 + tj * 16 + l15) * LB + lhalf * 8];
      b[tj].h[0] = *(const bf16x8*)(pb);
      b[tj].h[1] = *(const bf16x8*)(pb + 16);
    }
#pragma unroll
    for (int ti = 0; ti < 2; ++ti)
#pragma unroll
      for (int tj = 0; tj < 2; ++tj)
        acc[ti][tj] = __builtin_amdgcn_wmma_f32_16x16x32_bf16(
            false, a[ti].f, false, b[tj].f, (short)0, acc[ti][tj], false, false);
  };

  const int nkFull = K / BK;            // >= 1 for all our shapes
  const int rem    = K - nkFull * BK;   // 0 or 16 (K=400)

  // prologue: stage tile 0
  gload(sc8);
  sstore(0);
  __syncthreads();

  // steady state: load tile ks+1 (overlaps compute), compute ks, store, barrier
  for (int ks = 0; ks + 1 < nkFull; ++ks) {
    gload((ks + 1) * BK + sc8);
    compute(ks & 1);
    sstore((ks + 1) & 1);
    __syncthreads();
  }

  if (rem) {
    // guarded tail tile (chunk-aligned: all-or-nothing per bf16x8)
    const int gk = nkFull * BK + sc8;
    if (gk < K) {
      gload(gk);
    } else {
      bf16x8 zz = {};
      ra0 = zz; ra1 = zz; rb = zz;
    }
    compute((nkFull - 1) & 1);
    sstore(nkFull & 1);
    __syncthreads();
    compute(nkFull & 1);
  } else {
    compute((nkFull - 1) & 1);
  }

  // ---- epilogue: bias + activation + store ----
  // C/D layout: lanes 0-15 -> N=lane, M=r ; lanes 16-31 -> N=lane-16, M=r+8
#pragma unroll
  for (int tj = 0; tj < 2; ++tj) {
    int gn = blockN + wn * 32 + tj * 16 + l15;
    if (gn >= N) continue;
    float bv = bias[gn];
#pragma unroll
    for (int ti = 0; ti < 2; ++ti) {
#pragma unroll
      for (int r = 0; r < 8; ++r) {
        int gm = blockM + wm * 32 + ti * 16 + lhalf * 8 + r;
        float v = acc[ti][tj][r] + bv;
        size_t idx = (size_t)gm * ldc + gn;
        if constexpr (EPI == 0) {
          ((__bf16*)Cp)[idx] = f2bf(v > 0.f ? v : 0.f);
        } else if constexpr (EPI == 1) {
          ((float*)Cp)[idx] = 1.f / (1.f + __expf(-v));
        } else {
          ((float*)Cp)[idx] = v;
        }
      }
    }
  }
}

// ---------------------------------------------------------------------------
// rho = tanh(h1@W22 + b22), logs = h1@W23 + b23, std = sqrt(exp(logs))
// one wave (32 lanes) per row; dual dot over 400
// ---------------------------------------------------------------------------
__global__ __launch_bounds__(256) void rho_logs_kernel(
    const __bf16* __restrict__ h1,
    const float* __restrict__ W22, const float* __restrict__ b22,
    const float* __restrict__ W23, const float* __restrict__ b23,
    float* __restrict__ rho_out, float* __restrict__ logs_out,
    float* __restrict__ std_out)
{
  const int lane = threadIdx.x & 31;
  const int wave = threadIdx.x >> 5;
  const int row  = blockIdx.x * 8 + wave;
  const __bf16* hr = h1 + (size_t)row * HH;
  float s22 = 0.f, s23 = 0.f;
#pragma unroll
  for (int q = 0; q < 13; ++q) {
    int k = q * 32 + lane;
    if (k < HH) {
      float hv = bf2f(hr[k]);
      s22 = fmaf(hv, W22[k], s22);
      s23 = fmaf(hv, W23[k], s23);
    }
  }
#pragma unroll
  for (int off = 16; off > 0; off >>= 1) {
    s22 += __shfl_down(s22, off);
    s23 += __shfl_down(s23, off);
  }
  if (lane == 0) {
    float lg = s23 + b23[0];
    rho_out[row]  = tanhf(s22 + b22[0]);
    logs_out[row] = lg;
    std_out[row]  = sqrtf(__expf(lg));
  }
}

// ---------------------------------------------------------------------------
// z = cumsum(eps * std_row, axis=1) + mu    -> bf16
// one wave per row; each lane owns 16 contiguous of the 512 columns
// ---------------------------------------------------------------------------
__global__ __launch_bounds__(256) void cumsum_kernel(
    const float* __restrict__ eps, const float* __restrict__ std_arr,
    const float* __restrict__ mu, __bf16* __restrict__ z)
{
  const int lane = threadIdx.x & 31;
  const int wave = threadIdx.x >> 5;
  const int row  = blockIdx.x * 8 + wave;
  const float sd = std_arr[row];
  const float* er = eps + (size_t)row * ZD;
  const float* mr = mu  + (size_t)row * ZD;
  __bf16* zr = z + (size_t)row * ZD;
  const int j0 = lane * 16;

  float vals[16], muv[16];
  float lsum = 0.f;
#pragma unroll
  for (int q = 0; q < 4; ++q) {
    float4 e = *(const float4*)(er + j0 + q * 4);
    float4 m = *(const float4*)(mr + j0 + q * 4);
    vals[q*4+0] = e.x * sd; vals[q*4+1] = e.y * sd;
    vals[q*4+2] = e.z * sd; vals[q*4+3] = e.w * sd;
    muv[q*4+0] = m.x; muv[q*4+1] = m.y; muv[q*4+2] = m.z; muv[q*4+3] = m.w;
    lsum += vals[q*4+0] + vals[q*4+1] + vals[q*4+2] + vals[q*4+3];
  }
  // inclusive scan of lane partial sums -> exclusive prefix
  float s = lsum;
#pragma unroll
  for (int off = 1; off < 32; off <<= 1) {
    float t = __shfl_up(s, off);
    if (lane >= off) s += t;
  }
  float run = s - lsum;

  bf16x8 o0, o1;
#pragma unroll
  for (int j = 0; j < 16; ++j) {
    run += vals[j];
    __bf16 v = f2bf(run + muv[j]);
    if (j < 8) o0[j] = v; else o1[j - 8] = v;
  }
  *(bf16x8*)(zr + j0)     = o0;
  *(bf16x8*)(zr + j0 + 8) = o1;
}

// ---------------------------------------------------------------------------
extern "C" void kernel_launch(void* const* d_in, const int* in_sizes, int n_in,
                              void* d_out, int out_size, void* d_ws, size_t ws_size,
                              hipStream_t stream) {
  const float* x   = (const float*)d_in[0];
  const float* eps = (const float*)d_in[1];
  const float* W1  = (const float*)d_in[2];
  const float* b1  = (const float*)d_in[3];
  const float* W21 = (const float*)d_in[4];
  const float* b21 = (const float*)d_in[5];
  const float* W22 = (const float*)d_in[6];
  const float* b22 = (const float*)d_in[7];
  const float* W23 = (const float*)d_in[8];
  const float* b23 = (const float*)d_in[9];
  const float* W3  = (const float*)d_in[10];
  const float* b3  = (const float*)d_in[11];
  const float* W4  = (const float*)d_in[12];
  const float* b4  = (const float*)d_in[13];

  // Output: recon [B,D] ++ mu [B,Z] ++ rho [B] ++ logs [B] (all f32, flat)
  float* recon = (float*)d_out;
  float* mu    = recon + (size_t)NB * DD;
  float* rho   = mu    + (size_t)NB * ZD;
  float* logs  = rho   + NB;

  // Workspace layout (z/h3/std overlay the dead xb region after GEMM1):
  //   [0]           xb   bf16  134,217,728  (x in bf16; L2-resident at 128 MB)
  //   [0]           z    bf16   16,777,216  (overlays xb)
  //   [16,777,216]  h3   bf16   13,107,200  (overlays xb)
  //   [29,884,416]  std  f32        65,536  (overlays xb)
  //   [134,217,728] h1   bf16   13,107,200
  //   [147,324,928] W1T  bf16    3,276,800   [400,4096]
  //   [150,601,728] W21T bf16      409,600   [512,400]
  //   [151,011,328] W3T  bf16      409,600   [400,512]
  //   [151,420,928] W4T  bf16    3,276,800   [4096,400]
  char* ws = (char*)d_ws;
  __bf16* xb   = (__bf16*)(ws);
  __bf16* z    = (__bf16*)(ws);
  __bf16* h3   = (__bf16*)(ws + 16777216);
  float*  sdv  = (float*) (ws + 29884416);
  __bf16* h1   = (__bf16*)(ws + 134217728);
  __bf16* W1T  = (__bf16*)(ws + 147324928);
  __bf16* W21T = (__bf16*)(ws + 150601728);
  __bf16* W3T  = (__bf16*)(ws + 151011328);
  __bf16* W4T  = (__bf16*)(ws + 151420928);

  const dim3 blk(256);
  const int gM = NB / 128;              // 128

  // 0) x -> bf16 (makes the 7-pass GEMM1 A-stream L2-resident at 128 MB)
  cvt_bf16_kernel<<<dim3((NB * (size_t)DD) / 8 / 256), blk, 0, stream>>>(x, xb);

  // 0b) weights -> bf16, transposed to [N][K]
  transpose_cvt_kernel<<<dim3(HH / 16, DD / 16), blk, 0, stream>>>(W1,  W1T,  DD, HH);
  transpose_cvt_kernel<<<dim3(ZD / 16, HH / 16), blk, 0, stream>>>(W21, W21T, HH, ZD);
  transpose_cvt_kernel<<<dim3(HH / 16, ZD / 16), blk, 0, stream>>>(W3,  W3T,  ZD, HH);
  transpose_cvt_kernel<<<dim3(DD / 16, HH / 16), blk, 0, stream>>>(W4,  W4T,  HH, DD);

  // 1) h1 = relu(xb @ W1 + b1)          [16384,4096]x[4096,400]
  gemm_bias_act<0><<<dim3(gM, (HH + 63) / 64), blk, 0, stream>>>(
      xb, DD, W1T, b1, (void*)h1, HH, NB, HH, DD);

  // 2) rho/logs/std from h1
  rho_logs_kernel<<<dim3(NB / 8), blk, 0, stream>>>(
      h1, W22, b22, W23, b23, rho, logs, sdv);

  // 3) mu = h1 @ W21 + b21              [16384,400]x[400,512]
  gemm_bias_act<2><<<dim3(gM, ZD / 64), blk, 0, stream>>>(
      h1, HH, W21T, b21, (void*)mu, ZD, NB, ZD, HH);

  // 4) z = cumsum(eps*std) + mu  (bf16, overlays dead xb)
  cumsum_kernel<<<dim3(NB / 8), blk, 0, stream>>>(eps, sdv, mu, z);

  // 5) h3 = relu(z @ W3 + b3)           [16384,512]x[512,400]
  gemm_bias_act<0><<<dim3(gM, (HH + 63) / 64), blk, 0, stream>>>(
      z, ZD, W3T, b3, (void*)h3, HH, NB, HH, ZD);

  // 6) recon = sigmoid(h3 @ W4 + b4)    [16384,400]x[400,4096]
  gemm_bias_act<1><<<dim3(gM, DD / 64), blk, 0, stream>>>(
      h3, HH, W4T, b4, (void*)recon, DD, NB, DD, HH);
}
